// Encoder_40166534152835
// MI455X (gfx1250) — matrix operand order repaired and checked by reading, not compiled, not run on hardware
//
#include <hip/hip_runtime.h>
#include <cstdint>
#include <cstddef>

// ---------------------------------------------------------------------------
// CDNA5 (gfx1250) transformer encoder layer.
// f16 WMMA for all matmuls; all LDS tile staging via
// global_load_async_to_lds_b128 (ASYNCcnt); double-buffered GEMM and
// attention pipelines.  B=4, T=2048, D=1024, H=16, HD=64, FF=2048.  wave32.
// ---------------------------------------------------------------------------

typedef __attribute__((ext_vector_type(16))) _Float16 v16h;
typedef __attribute__((ext_vector_type(8)))  float    v8f;
typedef __attribute__((ext_vector_type(4)))  float    vf4;   // POD 16-byte chunk

union ABfrag { v16h v; vf4 f4[2]; _Float16 h[16]; };

constexpr int kB  = 4;
constexpr int kT  = 2048;
constexpr int kD  = 1024;
constexpr int kH  = 16;
constexpr int kHD = 64;
constexpr int kFF = 2048;
constexpr int kM  = kB * kT;        // 8192 token rows
constexpr int k3D = 3 * kD;         // fused QKV width

// ------------------------- CDNA5 async-to-LDS helpers ----------------------
// global_load_async_to_lds_b128: per-lane 16B global->LDS DMA, tracked by
// ASYNCcnt (ISA 10.x / 15.18.3).  GVS addressing: SADDR(64b SGPR) + VADDR(u32).

__device__ __forceinline__ uint32_t lds_u32(const void* p) {
  // flat->LDS: hardware uses addr[31:0]; shared aperture is 4GB aligned.
  return (uint32_t)(uintptr_t)p;
}

__device__ __forceinline__ void async_b128(uint32_t dst_lds, uint32_t src_off,
                                           const void* base) {
  asm volatile("global_load_async_to_lds_b128 %0, %1, %2"
               :: "v"(dst_lds), "v"(src_off), "s"(base) : "memory");
}

template <int N>
__device__ __forceinline__ void wait_asynccnt() {
  asm volatile("s_wait_asynccnt %0" :: "i"(N) : "memory");
}

// ---------------------------- small prep kernels ---------------------------

__global__ void cvt_f32_f16(const float* __restrict__ in, _Float16* __restrict__ out, int n) {
  int i = blockIdx.x * 256 + threadIdx.x;
  if (i < n) out[i] = (_Float16)in[i];
}

// out[n*K + k] = in[k*N + n]  (f32 -> f16, transpose to N-major weight layout)
__global__ void transpose_f32_f16(const float* __restrict__ in, _Float16* __restrict__ out,
                                  int K, int N) {
  int i = blockIdx.x * 256 + threadIdx.x;
  if (i >= K * N) return;
  int n = i / K, k = i - n * K;
  out[i] = (_Float16)in[(size_t)k * N + n];
}

// Wq/Wk/Wv are [H, D, HD]; pack transposed to [N=3D][K=D] f16 where output
// row (s*D + h*64 + kk) column d comes from W_s[h, d, kk].
__global__ void pack_wqkv_t(const float* __restrict__ Wq, const float* __restrict__ Wk,
                            const float* __restrict__ Wv, _Float16* __restrict__ out) {
  int i = blockIdx.x * 256 + threadIdx.x;
  if (i >= k3D * kD) return;
  int c  = i / kD;                    // output row: n in [0,3072)
  int d  = i - c * kD;                // k index
  int s  = c / kD;
  int hk = c - s * kD;                // h*64 + kk
  int h  = hk >> 6, kk = hk & 63;
  const float* W = (s == 0) ? Wq : (s == 1) ? Wk : Wv;
  out[i] = (_Float16)W[((size_t)h * kD + d) * kHD + kk];
}

__global__ void pack_bqkv(const float* __restrict__ bq, const float* __restrict__ bk,
                          const float* __restrict__ bv, float* __restrict__ out) {
  int i = blockIdx.x * 256 + threadIdx.x;
  if (i >= k3D) return;
  const float* src = (i < kD) ? bq : (i < 2 * kD) ? bk : bv;
  out[i] = src[i % kD];
}

// Tiled per-head V transpose: vt[((b*H + h)*HD + hd)*T + t] = V[b,t,h,hd].
// Both global sides are coalesced b128; LDS tile does the reshape.
// V is read exactly once per layer, so this one extra 32MB round trip is
// ~3us of HBM time and removes all VALU transposes from the attention loop.
__global__ __launch_bounds__(256) void transpose_v(const _Float16* __restrict__ qkv,
                                                   _Float16* __restrict__ vt) {
  __shared__ _Float16 tile[64][72];   // 64x64 tile, padded rows (144B, 16B-aligned)
  const int bh = blockIdx.y;
  const int b = bh >> 4, h = bh & 15;
  const int t0 = blockIdx.x * 64;
  const int tid = threadIdx.x;
  const int r = tid >> 2;             // 0..63
  const int c = (tid & 3) * 16;       // 0/16/32/48
  {
    const _Float16* p = qkv + (size_t)(b * kT + t0 + r) * k3D + 2 * kD + h * kHD + c;
    *(vf4*)&tile[r][c]     = *(const vf4*)p;
    *(vf4*)&tile[r][c + 8] = *(const vf4*)(p + 8);
  }
  __syncthreads();
  {
    ABfrag tmp;
    #pragma unroll
    for (int i = 0; i < 16; ++i) tmp.h[i] = tile[c + i][r];   // gather column r
    _Float16* q = vt + ((size_t)bh * kHD + r) * kT + t0 + c;
    *(vf4*)q       = tmp.f4[0];
    *(vf4*)(q + 8) = tmp.f4[1];
  }
}

// ------------------------------- WMMA GEMM ---------------------------------
// C[M,N] = A[M,K] @ Bt[N,K]^T + bias.
//   EPI=0: store f16        EPI=1: relu, store f16        EPI=2: +resid, f32
// Block 128x128, K-step 32, 8 waves, wave tile 32x64.  Double-buffered LDS
// fed entirely by global_load_async_to_lds_b128 (4 per thread per stage).

constexpr int BM = 128, BN = 128, BK = 32;

template <int EPI>
__global__ __launch_bounds__(256) void gemm_wmma(
    const _Float16* __restrict__ A, const _Float16* __restrict__ Bt,
    const float* __restrict__ bias, const float* __restrict__ resid,
    void* __restrict__ outp, int Mq, int N, int K) {
  __shared__ _Float16 sA[2][BM][BK];   // row-major A tiles   (2 x 8 KB)
  __shared__ _Float16 sB[2][BN][BK];   // N-major  B tiles    (2 x 8 KB)

  const int tid  = threadIdx.x;
  const int lane = tid & 31;
  const int wv   = tid >> 5;          // wave 0..7
  const int wm   = wv & 3;            // 4 row groups of 32
  const int wn   = wv >> 2;           // 2 col groups of 64
  const int l16  = lane & 15;
  const int hl   = lane >> 4;         // half-wave select
  const int bm0  = blockIdx.x * BM;
  const int bn0  = blockIdx.y * BN;
  (void)Mq;

  const int srow = tid >> 1;          // 0..127 staging row (A and B)
  const int scol = (tid & 1) * 16;    // 0 / 16

  auto issue_stage = [&](int k0, int buf) {
    uint32_t ga = (uint32_t)(((size_t)(bm0 + srow) * K + (size_t)(k0 + scol)) * sizeof(_Float16));
    async_b128(lds_u32(&sA[buf][srow][scol]),     ga,      A);
    async_b128(lds_u32(&sA[buf][srow][scol + 8]), ga + 16, A);
    uint32_t gb = (uint32_t)(((size_t)(bn0 + srow) * K + (size_t)(k0 + scol)) * sizeof(_Float16));
    async_b128(lds_u32(&sB[buf][srow][scol]),     gb,      Bt);
    async_b128(lds_u32(&sB[buf][srow][scol + 8]), gb + 16, Bt);
  };

  v8f acc[2][4] = {};
  const int nk = K / BK;

  issue_stage(0, 0);
  for (int i = 0; i < nk; ++i) {
    const int cur = i & 1;
    if (i + 1 < nk) {
      issue_stage((i + 1) * BK, cur ^ 1);
      wait_asynccnt<4>();             // stage i complete, stage i+1 in flight
    } else {
      wait_asynccnt<0>();
    }
    __syncthreads();

    // A frag (16-bit 16x32): lane M=l16; lanes<16: K=0..7,16..23; lanes>=16: +8
    ABfrag af[2], bf[4];
    #pragma unroll
    for (int mt = 0; mt < 2; ++mt) {
      const _Float16* p = &sA[cur][wm * 32 + mt * 16 + l16][hl * 8];
      af[mt].f4[0] = *(const vf4*)p;
      af[mt].f4[1] = *(const vf4*)(p + 16);
    }
    // B frag (32x16): lane N=l16; lanes<16 hold K=0..7,16..23; lanes>=16: +8
    #pragma unroll
    for (int nt = 0; nt < 4; ++nt) {
      const _Float16* p = &sB[cur][wn * 64 + nt * 16 + l16][hl * 8];
      bf[nt].f4[0] = *(const vf4*)p;
      bf[nt].f4[1] = *(const vf4*)(p + 16);
    }
    #pragma unroll
    for (int mt = 0; mt < 2; ++mt)
      #pragma unroll
      for (int nt = 0; nt < 4; ++nt)
        acc[mt][nt] = __builtin_amdgcn_wmma_f32_16x16x32_f16(
            false, af[mt].v, false, bf[nt].v, (short)0, acc[mt][nt], false, false);
    __syncthreads();
  }

  // C layout: vgpr r, lane l -> row = r + 8*(l>=16), col = l&15
  #pragma unroll
  for (int mt = 0; mt < 2; ++mt) {
    #pragma unroll
    for (int nt = 0; nt < 4; ++nt) {
      int cg = bn0 + wn * 64 + nt * 16 + l16;
      float bb = bias[cg];
      #pragma unroll
      for (int r = 0; r < 8; ++r) {
        int rg = bm0 + wm * 32 + mt * 16 + r + 8 * hl;
        size_t idx = (size_t)rg * N + cg;
        float vv = acc[mt][nt][r] + bb;
        if (EPI == 0)      ((_Float16*)outp)[idx] = (_Float16)vv;
        else if (EPI == 1) ((_Float16*)outp)[idx] = (_Float16)fmaxf(vv, 0.f);
        else               ((float*)outp)[idx]    = vv + resid[idx];
      }
    }
  }
}

// ---------------------------- flash attention ------------------------------
// grid = (T/64, B*H); 4 waves/WG, each wave = one 16-row query tile.
// 32-key blocks; K and pre-transposed V staged via async-to-LDS, double
// buffered; online softmax; WMMA for QK^T and PV.

__global__ __launch_bounds__(128) void attn_wmma(const _Float16* __restrict__ qkv,
                                                 const _Float16* __restrict__ vt,
                                                 _Float16* __restrict__ ctx) {
  const int bh = blockIdx.y;
  const int b = bh >> 4, h = bh & 15;
  const int rowBase = b * kT;
  const int qcol = h * kHD;
  const int kcol = kD + h * kHD;

  const int tid  = threadIdx.x;
  const int lane = tid & 31;
  const int wv   = tid >> 5;
  const int l16  = lane & 15;
  const int hl   = lane >> 4;
  const int q0   = blockIdx.x * 64 + wv * 16;    // local query row

  __shared__ _Float16 sK[2][32][64];   // [key][hd]  row-major (B-frag for QK^T)
  __shared__ _Float16 sV[2][64][32];   // [hd][key]  K-major   (B-frag for PV)
  __shared__ _Float16 sP[4][16][32];   // per-wave P staging (C-layout -> A-layout)

  // Q fragments (16x64 -> two 16x32 A-frags), resident for whole kernel
  ABfrag qf[2];
  {
    const _Float16* qp = qkv + (size_t)(rowBase + q0 + l16) * k3D + qcol;
    #pragma unroll
    for (int c = 0; c < 2; ++c) {
      qf[c].f4[0] = *(const vf4*)(qp + c * 32 + hl * 8);
      qf[c].f4[1] = *(const vf4*)(qp + c * 32 + hl * 8 + 16);
    }
  }

  v8f cacc[4] = {};
  float mrow[8], lrow[8];
  #pragma unroll
  for (int r = 0; r < 8; ++r) { mrow[r] = -1e30f; lrow[r] = 0.f; }

  const int kr = tid >> 2;             // K staging: key row 0..31
  const int kc = (tid & 3) * 16;       // K staging: hd col 0/16/32/48
  const int vr = tid >> 1;             // V staging: hd row 0..63
  const int vc = (tid & 1) * 16;       // V staging: key col 0/16
  const size_t vbase = (size_t)bh * kHD * kT;   // vt slab for this (b,h)

  auto issue_block = [&](int s0, int buf) {
    uint32_t gk = (uint32_t)(((size_t)(rowBase + s0 + kr) * k3D + (size_t)(kcol + kc))
                             * sizeof(_Float16));
    async_b128(lds_u32(&sK[buf][kr][kc]),     gk,      qkv);
    async_b128(lds_u32(&sK[buf][kr][kc + 8]), gk + 16, qkv);
    uint32_t gv = (uint32_t)((vbase + (size_t)vr * kT + (size_t)(s0 + vc))
                             * sizeof(_Float16));
    async_b128(lds_u32(&sV[buf][vr][vc]),     gv,      vt);
    async_b128(lds_u32(&sV[buf][vr][vc + 8]), gv + 16, vt);
  };

  constexpr int nblk = kT / 32;        // 64 key blocks
  issue_block(0, 0);
  for (int i = 0; i < nblk; ++i) {
    const int cur = i & 1;
    if (i + 1 < nblk) {
      issue_block((i + 1) * 32, cur ^ 1);
      wait_asynccnt<4>();             // block i landed; block i+1 in flight
    } else {
      wait_asynccnt<0>();
    }
    __syncthreads();                   // buffer `cur` ready for all waves

    // scores: two 16x16 key tiles, K-dim 64 = 2 chained WMMAs each
    v8f sc[2];
    #pragma unroll
    for (int j = 0; j < 2; ++j) {
      v8f a = {};
      #pragma unroll
      for (int c = 0; c < 2; ++c) {
        ABfrag bfr;
        const _Float16* p = &sK[cur][j * 16 + l16][c * 32 + hl * 8];
        bfr.f4[0] = *(const vf4*)p;
        bfr.f4[1] = *(const vf4*)(p + 16);
        a = __builtin_amdgcn_wmma_f32_16x16x32_f16(
                false, qf[c].v, false, bfr.v, (short)0, a, false, false);
      }
      sc[j] = a;
    }

    // online softmax (rows live in 16-lane halves; shfl_xor 1/2/4/8 reduces)
    constexpr float scale = 0.125f;   // 1/sqrt(64)
    float alpha[8];
    #pragma unroll
    for (int r = 0; r < 8; ++r) {
      float v0 = sc[0][r] * scale, v1 = sc[1][r] * scale;
      sc[0][r] = v0; sc[1][r] = v1;
      float v = fmaxf(v0, v1);
      #pragma unroll
      for (int off = 1; off < 16; off <<= 1) v = fmaxf(v, __shfl_xor(v, off, 32));
      float nm = fmaxf(mrow[r], v);
      alpha[r] = __expf(mrow[r] - nm);
      mrow[r] = nm;
    }
    #pragma unroll
    for (int r = 0; r < 8; ++r) {
      float p0 = __expf(sc[0][r] - mrow[r]);
      float p1 = __expf(sc[1][r] - mrow[r]);
      sP[wv][r + 8 * hl][l16]      = (_Float16)p0;   // C-layout -> LDS row-major
      sP[wv][r + 8 * hl][16 + l16] = (_Float16)p1;
      float s = p0 + p1;
      #pragma unroll
      for (int off = 1; off < 16; off <<= 1) s += __shfl_xor(s, off, 32);
      lrow[r] = lrow[r] * alpha[r] + s;
      #pragma unroll
      for (int nt = 0; nt < 4; ++nt) cacc[nt][r] *= alpha[r];
    }
    // sP is wave-private and LDS ops within a wave complete in order -> no
    // cross-wave barrier needed between the stores above and the loads below.

    // ctx += P(16x32) @ V(32x64): A-frag from sP, 4 B-frags from sV
    ABfrag pf;
    {
      const _Float16* p = &sP[wv][l16][hl * 8];
      pf.f4[0] = *(const vf4*)p;
      pf.f4[1] = *(const vf4*)(p + 16);
    }
    #pragma unroll
    for (int nt = 0; nt < 4; ++nt) {
      ABfrag vfr;
      const _Float16* p = &sV[cur][nt * 16 + l16][hl * 8];
      vfr.f4[0] = *(const vf4*)p;
      vfr.f4[1] = *(const vf4*)(p + 16);
      cacc[nt] = __builtin_amdgcn_wmma_f32_16x16x32_f16(
                     false, pf.v, false, vfr.v, (short)0, cacc[nt], false, false);
    }
    __syncthreads();                   // buffer `cur` free for re-issue at i+2
  }

  // normalize + store f16 context (concat-head layout [row][h*64 + c])
  #pragma unroll
  for (int nt = 0; nt < 4; ++nt) {
    #pragma unroll
    for (int r = 0; r < 8; ++r) {
      int m = q0 + r + 8 * hl;
      float v = cacc[nt][r] / lrow[r];
      ctx[(size_t)(rowBase + m) * kD + h * kHD + nt * 16 + l16] = (_Float16)v;
    }
  }
}

// ------------------------------- LayerNorm ---------------------------------

__global__ __launch_bounds__(256) void ln_kernel(const float* __restrict__ in,
                                                 const float* __restrict__ g,
                                                 const float* __restrict__ bta,
                                                 float* __restrict__ out,
                                                 _Float16* __restrict__ outH, int Dn) {
  int row = blockIdx.x;
  const float* xr = in + (size_t)row * Dn;
  float s = 0.f, s2 = 0.f;
  for (int i = threadIdx.x; i < Dn; i += 256) { float v = xr[i]; s += v; s2 += v * v; }
  #pragma unroll
  for (int off = 16; off >= 1; off >>= 1) {
    s  += __shfl_xor(s,  off, 32);
    s2 += __shfl_xor(s2, off, 32);
  }
  __shared__ float rs[8], rs2[8];
  __shared__ float mu_s, rsig_s;
  int wv = threadIdx.x >> 5;
  if ((threadIdx.x & 31) == 0) { rs[wv] = s; rs2[wv] = s2; }
  __syncthreads();
  if (threadIdx.x == 0) {
    float a = 0.f, a2 = 0.f;
    for (int i = 0; i < 8; ++i) { a += rs[i]; a2 += rs2[i]; }
    float mu = a / Dn;
    float var = a2 / Dn - mu * mu;
    mu_s = mu; rsig_s = rsqrtf(var + 1e-5f);
  }
  __syncthreads();
  float mu = mu_s, rsg = rsig_s;
  for (int i = threadIdx.x; i < Dn; i += 256) {
    float v = (xr[i] - mu) * rsg * g[i] + bta[i];
    out[(size_t)row * Dn + i] = v;
    if (outH) outH[(size_t)row * Dn + i] = (_Float16)v;
  }
}

// ------------------------------- launcher ----------------------------------

extern "C" void kernel_launch(void* const* d_in, const int* in_sizes, int n_in,
                              void* d_out, int out_size, void* d_ws, size_t ws_size,
                              hipStream_t stream) {
  const float* x   = (const float*)d_in[0];
  const float* Wq  = (const float*)d_in[1];
  const float* bq  = (const float*)d_in[2];
  const float* Wk  = (const float*)d_in[3];
  const float* bk  = (const float*)d_in[4];
  const float* Wv  = (const float*)d_in[5];
  const float* bv  = (const float*)d_in[6];
  const float* Wo  = (const float*)d_in[7];
  const float* bo  = (const float*)d_in[8];
  const float* g1  = (const float*)d_in[9];
  const float* be1 = (const float*)d_in[10];
  const float* W1  = (const float*)d_in[11];
  const float* b1  = (const float*)d_in[12];
  const float* W2  = (const float*)d_in[13];
  const float* b2  = (const float*)d_in[14];
  const float* g2  = (const float*)d_in[15];
  const float* be2 = (const float*)d_in[16];
  (void)in_sizes; (void)n_in; (void)out_size; (void)ws_size;

  uint8_t* ws = (uint8_t*)d_ws;
  size_t off = 0;
  auto alloc = [&](size_t bytes) -> void* {
    void* p = ws + off;
    off += (bytes + 255) & ~(size_t)255;
    return p;
  };

  _Float16* xh    = (_Float16*)alloc((size_t)kM * kD * 2);   // later reused as x1h
  _Float16* wqkvt = (_Float16*)alloc((size_t)k3D * kD * 2);  // [3D][D]
  float*    bqkv  = (float*)   alloc((size_t)k3D * 4);
  _Float16* wot   = (_Float16*)alloc((size_t)kD * kD * 2);   // [D][D]   (Wo^T)
  _Float16* w1t   = (_Float16*)alloc((size_t)kFF * kD * 2);  // [FF][D]  (W1^T)
  _Float16* w2t   = (_Float16*)alloc((size_t)kD * kFF * 2);  // [D][FF]  (W2^T)
  _Float16* qkvh  = (_Float16*)alloc((size_t)kM * k3D * 2);  // later reused as h1h
  _Float16* vth   = (_Float16*)alloc((size_t)kM * kD * 2);   // per-head V^T
  _Float16* ctxh  = (_Float16*)alloc((size_t)kM * kD * 2);
  float*    t0    = (float*)   alloc((size_t)kM * kD * 4);   // attn resid sum; reused as y2
  float*    x1    = (float*)   alloc((size_t)kM * kD * 4);

  _Float16* x1h = xh;     // xh dead after QKV GEMM
  _Float16* h1h = qkvh;   // qkvh dead after attention
  float*    y2  = t0;     // t0 dead after LN1

  // f32 -> f16 conversions / weight packing (transposed to [N][K])
  {
    int n = kM * kD;
    cvt_f32_f16<<<dim3((n + 255) / 256), dim3(256), 0, stream>>>(x, xh, n);
  }
  pack_wqkv_t<<<dim3((k3D * kD + 255) / 256), dim3(256), 0, stream>>>(Wq, Wk, Wv, wqkvt);
  pack_bqkv<<<dim3((k3D + 255) / 256), dim3(256), 0, stream>>>(bq, bk, bv, bqkv);
  transpose_f32_f16<<<dim3((kD * kD + 255) / 256), dim3(256), 0, stream>>>(Wo, wot, kD, kD);
  transpose_f32_f16<<<dim3((kD * kFF + 255) / 256), dim3(256), 0, stream>>>(W1, w1t, kD, kFF);
  transpose_f32_f16<<<dim3((kFF * kD + 255) / 256), dim3(256), 0, stream>>>(W2, w2t, kFF, kD);

  // 1) fused QKV projection: [8192,1024] x [1024,3072] -> f16
  gemm_wmma<0><<<dim3(kM / BM, k3D / BN), dim3(256), 0, stream>>>(
      xh, wqkvt, bqkv, nullptr, qkvh, kM, k3D, kD);
  // 1b) per-head V transpose -> [B][H][HD][T]
  transpose_v<<<dim3(kT / 64, kB * kH), dim3(256), 0, stream>>>(qkvh, vth);
  // 2) flash attention -> ctx f16 [8192,1024]
  attn_wmma<<<dim3(kT / 64, kB * kH), dim3(128), 0, stream>>>(qkvh, vth, ctxh);
  // 3) output projection + residual (f32)
  gemm_wmma<2><<<dim3(kM / BM, kD / BN), dim3(256), 0, stream>>>(
      ctxh, wot, bo, x, t0, kM, kD, kD);
  // 4) LayerNorm1 -> x1 (f32) + x1h (f16)
  ln_kernel<<<dim3(kM), dim3(256), 0, stream>>>(t0, g1, be1, x1, x1h, kD);
  // 5) FFN1 + ReLU -> h1 f16 [8192,2048]
  gemm_wmma<1><<<dim3(kM / BM, kFF / BN), dim3(256), 0, stream>>>(
      x1h, w1t, b1, nullptr, h1h, kM, kFF, kD);
  // 6) FFN2 + residual (f32)
  gemm_wmma<2><<<dim3(kM / BM, kD / BN), dim3(256), 0, stream>>>(
      h1h, w2t, b2, x1, y2, kM, kD, kFF);
  // 7) LayerNorm2 -> d_out (f32)
  ln_kernel<<<dim3(kM), dim3(256), 0, stream>>>(y2, g2, be2, (float*)d_out, nullptr, kD);
}
